// DeepLabHeadV3Plus_duq_21131239096396
// MI455X (gfx1250) — matrix operand order repaired and checked by reading, not compiled, not run on hardware
//
#include <hip/hip_runtime.h>
#include <hip/hip_bf16.h>

// ---------- types & helpers ----------
typedef __attribute__((ext_vector_type(16))) __bf16        v16bf;
typedef __attribute__((ext_vector_type(8)))  float         v8f;
typedef __attribute__((ext_vector_type(4)))  unsigned int  v4u;

// explicit global-address-space pointers (force global_load, not flat_load)
typedef const __attribute__((address_space(1))) unsigned short* gus_p;
typedef const __attribute__((address_space(1))) v4u*            gv4u_p;

union BfTile { v16bf v; unsigned int u[8]; v4u q[2]; };

__device__ __forceinline__ unsigned int f2bf_bits(float f) {
    unsigned int u = __builtin_bit_cast(unsigned int, f);
    unsigned int r = (u + 0x7FFFu + ((u >> 16) & 1u)) >> 16;  // RNE bf16
    return r & 0xFFFFu;
}
__device__ __forceinline__ unsigned short f2bf(float f) { return (unsigned short)f2bf_bits(f); }
__device__ __forceinline__ float bf2f(unsigned short b) {
    return __builtin_bit_cast(float, ((unsigned int)b) << 16);
}
__device__ __forceinline__ unsigned int pack2bf(float f0, float f1) {
    return f2bf_bits(f0) | (f2bf_bits(f1) << 16);
}

// WMMA bf16 fragment K-pair origins (this code's contiguous k = 16h + s order, A and B alike)
__device__ __forceinline__ int a_k0(int v, int h) { return 2 * v + 16 * h; }
__device__ __forceinline__ int b_k0(int v, int h) { return 2 * v + 16 * h; }

// ---------- weight pre-packing into WMMA A-fragment layout ----------
// Ap[u32 i], i = ((mtile*(taps*nkb) + tap*nkb + kb32)*32 + lane)*8 + v ; tap = ky*3+kx row-major
__global__ void pack_wA(const float* __restrict__ Wsrc, unsigned int* __restrict__ Ap,
                        int KperTap, int CinW, int taps, int total) {
    int i = blockIdx.x * blockDim.x + threadIdx.x;
    if (i >= total) return;
    int v    = i & 7;
    int lane = (i >> 3) & 31;
    int rest = i >> 8;
    int nkb  = KperTap >> 5;
    int nkbT = taps * nkb;
    int kbG  = rest % nkbT;
    int mt   = rest / nkbT;
    int tap  = kbG / nkb;
    int kb   = (kbG % nkb) << 5;
    int hl = lane >> 4, lrow = lane & 15;
    int co = mt * 16 + lrow;
    int ka = kb + a_k0(v, hl);
    float f0 = (ka < CinW)     ? Wsrc[((size_t)co * CinW + ka) * taps + tap]     : 0.f;
    float f1 = (ka + 1 < CinW) ? Wsrc[((size_t)co * CinW + ka + 1) * taps + tap] : 0.f;
    Ap[i] = pack2bf(f0, f1);
}

// DUQ W [256,19,128] -> B-fragment pack: i = (((n*16 + mtile)*4 + kb)*32 + lane)*8 + v
__global__ void pack_wB_duq(const float* __restrict__ Wp, unsigned int* __restrict__ Bp, int total) {
    int i = blockIdx.x * blockDim.x + threadIdx.x;
    if (i >= total) return;
    int v    = i & 7;
    int lane = (i >> 3) & 31;
    int kb   = (i >> 8) & 3;
    int mt   = (i >> 10) & 15;
    int n    = i >> 14;
    int hl = lane >> 4, lrow = lane & 15;
    int m = mt * 16 + lrow;
    int k = kb * 32 + b_k0(v, hl);
    const float* wr = Wp + ((size_t)m * 19 + n) * 128 + k;
    Bp[i] = pack2bf(wr[0], wr[1]);
}

// ---------- elementwise / small kernels ----------
__global__ void nchw2nhwc_bf16(const float* __restrict__ in, unsigned short* __restrict__ dst,
                               int C, int Npix, int total) {
    int i = blockIdx.x * blockDim.x + threadIdx.x;
    if (i >= total) return;
    int c   = i % C;
    int pix = (i / C) % Npix;
    int b   = i / (C * Npix);
    dst[i] = f2bf(in[((size_t)b * C + c) * Npix + pix]);
}

__global__ void gp_kernel(const float* __restrict__ outF, float* __restrict__ gp) {
    int t = blockIdx.x * blockDim.x + threadIdx.x;
    if (t >= 2 * 2048) return;
    const float* p = outF + (size_t)t * 512;
    float s = 0.f;
    for (int i = 0; i < 512; ++i) s += p[i];
    gp[t] = s * (1.0f / 512.0f);
}

__global__ void b5_kernel(const float* __restrict__ apw, const float* __restrict__ gp,
                          unsigned short* __restrict__ cat5) {
    int b  = blockIdx.x;
    int co = threadIdx.x;
    const float* w = apw + (size_t)co * 2048;
    const float* g = gp + (size_t)b * 2048;
    float s = 0.f;
    for (int ci = 0; ci < 2048; ++ci) s += w[ci] * g[ci];
    s = fmaxf(s, 0.f);
    unsigned short bs = f2bf(s);
    for (int p = 0; p < 512; ++p)
        cat5[((size_t)(b * 512 + p)) * 1280 + 1024 + co] = bs;
}

// zero pad channels of feat + the shared zero row used for OOB conv taps
__global__ void zero_init_kernel(unsigned short* __restrict__ feat, unsigned short* __restrict__ zbuf) {
    int i = blockIdx.x * blockDim.x + threadIdx.x;
    if (i < 2048) zbuf[i] = 0;
    if (i >= 2 * 8192 * 16) return;
    int j   = i % 16;
    int pix = (i / 16) % 8192;
    int b   = i / (16 * 8192);
    feat[((size_t)(b * 8192 + pix)) * 320 + 304 + j] = 0;
}

__global__ void upsample_kernel(const unsigned short* __restrict__ asppT,
                                unsigned short* __restrict__ feat) {
    int i = blockIdx.x * blockDim.x + threadIdx.x;
    if (i >= 2 * 8192 * 256) return;
    int c    = i % 256;
    int opix = (i / 256) % 8192;
    int b    = i / (256 * 8192);
    int oy = opix / 128, ox = opix % 128;
    float sy = oy * 0.25f - 0.375f;
    float sx = ox * 0.25f - 0.375f;
    float fy0 = floorf(sy), fx0 = floorf(sx);
    float wy = sy - fy0, wx = sx - fx0;
    int y0 = (int)fy0, x0 = (int)fx0;
    int y1 = y0 + 1, x1 = x0 + 1;
    y0 = min(max(y0, 0), 15); y1 = min(max(y1, 0), 15);
    x0 = min(max(x0, 0), 31); x1 = min(max(x1, 0), 31);
    const unsigned short* base = asppT + (size_t)b * 512 * 256;
    float v00 = bf2f(base[(y0 * 32 + x0) * 256 + c]);
    float v01 = bf2f(base[(y0 * 32 + x1) * 256 + c]);
    float v10 = bf2f(base[(y1 * 32 + x0) * 256 + c]);
    float v11 = bf2f(base[(y1 * 32 + x1) * 256 + c]);
    float v = (1.f - wy) * ((1.f - wx) * v00 + wx * v01) + wy * ((1.f - wx) * v10 + wx * v11);
    feat[((size_t)(b * 8192 + opix)) * 320 + 48 + c] = f2bf(v);
}

// ---------- implicit-GEMM conv (WMMA bf16, 16x64 tile per wave) ----------
#define NT 4
__global__ void conv_gemm(const v4u* __restrict__ Apq, const float* __restrict__ bias,
                          const unsigned short* __restrict__ act,
                          const unsigned short* __restrict__ zbuf,
                          unsigned short* __restrict__ dstBf, float* __restrict__ dstF,
                          int M, int KperTap, int taps, int dil,
                          int Hs, int Wd, int actC, int chOff, int destC, int doRelu) {
    const int Npix = Hs * Wd;
    const int lane = threadIdx.x;
    const int hl   = lane >> 4;
    const int lrow = lane & 15;
    const int bz   = blockIdx.z;

    int n[NT], py[NT], px[NT];
#pragma unroll
    for (int t = 0; t < NT; ++t) {
        n[t]  = blockIdx.y * (16 * NT) + t * 16 + lrow;
        py[t] = n[t] / Wd;
        px[t] = n[t] % Wd;
    }

    v8f acc[NT];
#pragma unroll
    for (int t = 0; t < NT; ++t) acc[t] = (v8f){0.f, 0.f, 0.f, 0.f, 0.f, 0.f, 0.f, 0.f};

    const ptrdiff_t baseB = (ptrdiff_t)bz * Npix * actC;

    // A fragment stream: 64 v4u per k-block (32 lanes x 2)
    const v4u* Ab = Apq + (size_t)blockIdx.x * taps * (KperTap >> 5) * 64 + lane * 2;

    const int r = (taps == 9) ? 1 : 0;   // tap radius (pack order is ky,kx row-major)
    for (int ky = -r; ky <= r; ++ky) {
        for (int kx = -r; kx <= r; ++kx) {
            gus_p bp[NT];   // explicit AS1 (global) pointers
#pragma unroll
            for (int t = 0; t < NT; ++t) {
                int sy = py[t] + ky * dil, sx = px[t] + kx * dil;
                bool vld = (sy >= 0) & (sy < Hs) & (sx >= 0) & (sx < Wd);
                // OOB lanes read the zeroed scratch row (branchless)
                const unsigned short* p =
                    vld ? (act + baseB + (ptrdiff_t)(sy * Wd + sx) * actC + 16 * hl)
                        : (zbuf + 16 * hl);
                bp[t] = (gus_p)(uintptr_t)p;
            }
#pragma unroll 2
            for (int kb = 0; kb < KperTap; kb += 32) {
                BfTile A;
                A.q[0] = Ab[0];
                A.q[1] = Ab[1];
                __builtin_prefetch((const void*)(Ab + 256), 0, 1);
                Ab += 64;
#pragma unroll
                for (int t = 0; t < NT; ++t) {
                    BfTile B;
                    B.q[0] = *(gv4u_p)(bp[t]);
                    B.q[1] = *(gv4u_p)(bp[t] + 8);
                    bp[t] += 32;
                    acc[t] = __builtin_amdgcn_wmma_f32_16x16x32_bf16(false, A.v, false, B.v,
                                                                     (short)0, acc[t], false, false);
                }
            }
        }
    }

    const int m0 = blockIdx.x * 16 + 8 * hl;
#pragma unroll
    for (int t = 0; t < NT; ++t) {
        float vals[8];
#pragma unroll
        for (int v = 0; v < 8; ++v) {
            float x = acc[t][v];
            if (bias) x += bias[m0 + v];
            if (doRelu) x = fmaxf(x, 0.f);
            vals[v] = x;
        }
        if (dstBf) {
            unsigned int* dp32 = (unsigned int*)(dstBf + ((size_t)bz * Npix + n[t]) * destC + chOff + m0);
#pragma unroll
            for (int v = 0; v < 4; ++v) dp32[v] = pack2bf(vals[2 * v], vals[2 * v + 1]);
        }
        if (dstF) {
#pragma unroll
            for (int v = 0; v < 8; ++v)
                dstF[((size_t)bz * M + (m0 + v)) * Npix + n[t]] = vals[v];
        }
    }
}

// ---------- DUQ RBF (WMMA bf16) ----------
__global__ void duq_kernel(const unsigned short* __restrict__ zt, const v4u* __restrict__ Bpq,
                           const float* __restrict__ mp, const float* __restrict__ Nbuf,
                           float* __restrict__ ypred) {
    const int ptile = blockIdx.x;   // 0..1023
    const int n     = blockIdx.y;   // 0..18
    const int lane  = threadIdx.x;
    const int hl    = lane >> 4;
    const int lrow  = lane & 15;

    BfTile A[4];
    const unsigned short* zrow = zt + (size_t)(ptile * 16 + lrow) * 128;
#pragma unroll
    for (int kb = 0; kb < 4; ++kb) {
        A[kb].q[0] = *(gv4u_p)(uintptr_t)(zrow + kb * 32 + 16 * hl);
        A[kb].q[1] = *(gv4u_p)(uintptr_t)(zrow + kb * 32 + 16 * hl + 8);
    }

    const float invN = 1.0f / Nbuf[n];
    float acc[8] = {0.f, 0.f, 0.f, 0.f, 0.f, 0.f, 0.f, 0.f};

    const v4u* Bb = Bpq + (size_t)n * 64 * 64 + lane * 2;
    for (int mtile = 0; mtile < 16; ++mtile) {
        const int mcol = mtile * 16 + lrow;
        v8f c = {};
#pragma unroll
        for (int kb = 0; kb < 4; ++kb) {
            BfTile B;
            B.q[0] = Bb[0];
            B.q[1] = Bb[1];
            Bb += 64;
            c = __builtin_amdgcn_wmma_f32_16x16x32_bf16(false, A[kb].v, false, B.v,
                                                        (short)0, c, false, false);
        }
        float emb = mp[mcol * 19 + n] * invN;
#pragma unroll
        for (int v = 0; v < 8; ++v) {
            float d = c[v] - emb;
            acc[v] += d * d;
        }
    }
#pragma unroll
    for (int mask = 1; mask < 16; mask <<= 1)
#pragma unroll
        for (int v = 0; v < 8; ++v)
            acc[v] += __shfl_xor(acc[v], mask, 32);

    if (lrow == 0) {
#pragma unroll
        for (int v = 0; v < 8; ++v) {
            int p = ptile * 16 + v + 8 * hl;
            int b = p >> 13, pix = p & 8191;
            ypred[(size_t)b * 19 * 8192 + (size_t)n * 8192 + pix] = expf(-acc[v] * (1.0f / 512.0f));
        }
    }
}

// ---------- host ----------
extern "C" void kernel_launch(void* const* d_in, const int* in_sizes, int n_in,
                              void* d_out, int out_size, void* d_ws, size_t ws_size,
                              hipStream_t stream) {
    const float* low_level = (const float*)d_in[0];
    const float* outF      = (const float*)d_in[1];
    const float* proj_w    = (const float*)d_in[2];
    const float* a1_w      = (const float*)d_in[3];
    const float* a2_w      = (const float*)d_in[4];
    const float* a3_w      = (const float*)d_in[5];
    const float* a4_w      = (const float*)d_in[6];
    const float* apool_w   = (const float*)d_in[7];
    const float* aproj_w   = (const float*)d_in[8];
    const float* cls3_w    = (const float*)d_in[9];
    const float* cls1_w    = (const float*)d_in[10];
    const float* cls1_b    = (const float*)d_in[11];
    const float* Wp        = (const float*)d_in[12];
    const float* mp        = (const float*)d_in[13];
    const float* Nbuf      = (const float*)d_in[14];

    float* ypred = (float*)d_out;                       // [2,19,64,128]
    float* zf32  = (float*)d_out + 2 * 19 * 64 * 128;   // [2,128,64,128]

    char* ws = (char*)d_ws;
    size_t off = 0;
    auto carve = [&](size_t bytes) { void* p = ws + off; off += (bytes + 255) & ~(size_t)255; return p; };
    unsigned short* outT  = (unsigned short*)carve((size_t)2 * 512 * 2048 * 2);
    unsigned short* lowT  = (unsigned short*)carve((size_t)2 * 8192 * 256 * 2);
    unsigned short* cat5  = (unsigned short*)carve((size_t)2 * 512 * 1280 * 2);
    unsigned short* asppT = (unsigned short*)carve((size_t)2 * 512 * 256 * 2);
    unsigned short* feat  = (unsigned short*)carve((size_t)2 * 8192 * 320 * 2);
    unsigned short* hT    = (unsigned short*)carve((size_t)2 * 8192 * 256 * 2);
    unsigned short* zt    = (unsigned short*)carve((size_t)2 * 8192 * 128 * 2);
    float*          gp    = (float*)carve((size_t)2 * 2048 * 4);
    unsigned short* zbuf  = (unsigned short*)carve((size_t)2048 * 2);   // shared OOB zero row
    // packed weights (u32 counts = M*taps*KperTap/2)
    unsigned int* projA  = (unsigned int*)carve((size_t)6144 * 4);
    unsigned int* a1A    = (unsigned int*)carve((size_t)262144 * 4);
    unsigned int* a2A    = (unsigned int*)carve((size_t)2359296 * 4);
    unsigned int* a3A    = (unsigned int*)carve((size_t)2359296 * 4);
    unsigned int* a4A    = (unsigned int*)carve((size_t)2359296 * 4);
    unsigned int* aprojA = (unsigned int*)carve((size_t)163840 * 4);
    unsigned int* cls3A  = (unsigned int*)carve((size_t)368640 * 4);
    unsigned int* cls1A  = (unsigned int*)carve((size_t)16384 * 4);
    unsigned int* duqB   = (unsigned int*)carve((size_t)311296 * 4);

    auto grid1 = [](int tot) { return (tot + 255) / 256; };

    // weight packing (fragment layout, bf16)
    pack_wA<<<grid1(6144),    256, 0, stream>>>(proj_w, projA, 256, 256, 1, 6144);
    pack_wA<<<grid1(262144),  256, 0, stream>>>(a1_w,   a1A, 2048, 2048, 1, 262144);
    pack_wA<<<grid1(2359296), 256, 0, stream>>>(a2_w,   a2A, 2048, 2048, 9, 2359296);
    pack_wA<<<grid1(2359296), 256, 0, stream>>>(a3_w,   a3A, 2048, 2048, 9, 2359296);
    pack_wA<<<grid1(2359296), 256, 0, stream>>>(a4_w,   a4A, 2048, 2048, 9, 2359296);
    pack_wA<<<grid1(163840),  256, 0, stream>>>(aproj_w, aprojA, 1280, 1280, 1, 163840);
    pack_wA<<<grid1(368640),  256, 0, stream>>>(cls3_w, cls3A, 320, 304, 9, 368640);
    pack_wA<<<grid1(16384),   256, 0, stream>>>(cls1_w, cls1A, 256, 256, 1, 16384);
    pack_wB_duq<<<grid1(311296), 256, 0, stream>>>(Wp, duqB, 311296);

    // layout transposes to NHWC bf16
    nchw2nhwc_bf16<<<grid1(2 * 512 * 2048), 256, 0, stream>>>(outF, outT, 2048, 512, 2 * 512 * 2048);
    nchw2nhwc_bf16<<<grid1(2 * 8192 * 256), 256, 0, stream>>>(low_level, lowT, 256, 8192, 2 * 8192 * 256);

    gp_kernel<<<grid1(2 * 2048), 256, 0, stream>>>(outF, gp);
    b5_kernel<<<2, 256, 0, stream>>>(apool_w, gp, cat5);
    zero_init_kernel<<<grid1(2 * 8192 * 16), 256, 0, stream>>>(feat, zbuf);

    // proj: 1x1 256->48 on low -> feat ch[0..48)
    conv_gemm<<<dim3(3, 128, 2), 32, 0, stream>>>((const v4u*)projA, nullptr, lowT, zbuf, feat, nullptr,
                                                  48, 256, 1, 1, 64, 128, 256, 0, 320, 1);
    // ASPP branches -> cat5
    conv_gemm<<<dim3(16, 8, 2), 32, 0, stream>>>((const v4u*)a1A, nullptr, outT, zbuf, cat5, nullptr,
                                                 256, 2048, 1, 1, 16, 32, 2048, 0, 1280, 1);
    conv_gemm<<<dim3(16, 8, 2), 32, 0, stream>>>((const v4u*)a2A, nullptr, outT, zbuf, cat5, nullptr,
                                                 256, 2048, 9, 12, 16, 32, 2048, 256, 1280, 1);
    conv_gemm<<<dim3(16, 8, 2), 32, 0, stream>>>((const v4u*)a3A, nullptr, outT, zbuf, cat5, nullptr,
                                                 256, 2048, 9, 24, 16, 32, 2048, 512, 1280, 1);
    conv_gemm<<<dim3(16, 8, 2), 32, 0, stream>>>((const v4u*)a4A, nullptr, outT, zbuf, cat5, nullptr,
                                                 256, 2048, 9, 36, 16, 32, 2048, 768, 1280, 1);
    // aspp projection: 1x1 1280->256
    conv_gemm<<<dim3(16, 8, 2), 32, 0, stream>>>((const v4u*)aprojA, nullptr, cat5, zbuf, asppT, nullptr,
                                                 256, 1280, 1, 1, 16, 32, 1280, 0, 256, 1);
    upsample_kernel<<<grid1(2 * 8192 * 256), 256, 0, stream>>>(asppT, feat);
    // classifier 3x3 (K padded 304->320) -> hT
    conv_gemm<<<dim3(16, 128, 2), 32, 0, stream>>>((const v4u*)cls3A, nullptr, feat, zbuf, hT, nullptr,
                                                   256, 320, 9, 1, 64, 128, 320, 0, 256, 1);
    // classifier 1x1 + bias -> z (f32 NCHW) and zt (bf16 NHWC)
    conv_gemm<<<dim3(8, 128, 2), 32, 0, stream>>>((const v4u*)cls1A, cls1_b, hT, zbuf, zt, zf32,
                                                  128, 256, 1, 1, 64, 128, 256, 0, 128, 0);
    // DUQ RBF -> y_pred
    duq_kernel<<<dim3(1024, 19), 32, 0, stream>>>(zt, (const v4u*)duqB, mp, Nbuf, ypred);
}
